// EdgeConvBlock_35287451304393
// MI455X (gfx1250) — matrix operand order repaired and checked by reading, not compiled, not run on hardware
//
#include <hip/hip_runtime.h>
#include <math.h>

// ---------------- CDNA5 WMMA vector types ----------------
typedef __attribute__((ext_vector_type(16))) __bf16 v16bf;
typedef __attribute__((ext_vector_type(8)))  __bf16 v8bf;
typedef __attribute__((ext_vector_type(8)))  float  v8f;

// ---------------- problem dims ----------------
static constexpr int BATCH = 32;
static constexpr int CIN   = 64;
static constexpr int NPTS  = 128;   // N; K = N-1 = 127 edges, padded to 128

// LDS row strides (bf16 elements). 136*2=272B and 72*2=144B are 16B multiples
// so ds_load/store_b128 stay aligned; odd 16B-block counts avoid the worst
// bank-conflict patterns on column walks.
static constexpr int ASTR = 136;    // 128-channel buffer (Xcat / Y2)
static constexpr int BSTR = 72;     // 64-channel buffer  (Y0/Y1/att-hidden)

// ---------------- d_ws layout (bytes) ----------------
static constexpr int W0_OFF  = 0;               // 64x128 bf16
static constexpr int W1_OFF  = 16384;           // 64x64  bf16
static constexpr int W2_OFF  = 24576;           // 128x64 bf16
static constexpr int WA0_OFF = 40960;           // 64x128 bf16
static constexpr int S0_OFF  = 57344;           // 64 f32 folded BN scale
static constexpr int B0_OFF  = S0_OFF  + 256;   // 64 f32 folded BN bias
static constexpr int S1_OFF  = B0_OFF  + 256;
static constexpr int B1_OFF  = S1_OFF  + 256;
static constexpr int S2_OFF  = B1_OFF  + 256;   // 128 f32
static constexpr int B2_OFF  = S2_OFF  + 512;
static constexpr int SA_OFF  = B2_OFF  + 512;   // 64 f32
static constexpr int BA_OFF  = SA_OFF  + 256;
static constexpr int SSC_OFF = BA_OFF  + 256;   // 128 f32
static constexpr int BSC_OFF = SSC_OFF + 512;

// ---------------- WMMA fragment helpers ----------------
// A tile (16x32 bf16): lane L -> row m0+ (L&15); elems 0..15 = K kt*32+b8..+15,
// b8 = (L>=16)?16:0  (contiguous order, matches load_btile).
__device__ __forceinline__ v16bf load_atile(const __bf16* __restrict__ W, int C,
                                            int m0c, int kt, int lane) {
  const int o  = m0c + (lane & 15);
  const int b8 = (lane >> 4) << 4;
  const __bf16* base = W + o * C + kt * 32 + b8;
  union { v16bf v; v8bf h[2]; } u;
  u.h[0] = *(const v8bf*)(base);
  u.h[1] = *(const v8bf*)(base + 8);
  return u.v;
}

// B tile (32x16 bf16): lane L -> edge col nt*16+(L&15); elems = 16 consecutive
// channels starting at ck + ((L>=16)?16:0). LDS layout is [edge][channel].
__device__ __forceinline__ v16bf load_btile(const __bf16* __restrict__ lds, int stride,
                                            int nt, int ck, int lane) {
  const int col = nt * 16 + (lane & 15);
  const int c   = ck + ((lane >> 4) << 4);
  const v8bf* p = (const v8bf*)(lds + col * stride + c);
  union { v16bf v; v8bf h[2]; } u;
  u.h[0] = p[0];
  u.h[1] = p[1];
  return u.v;
}

// D tile (16x16 f32): VGPR v -> channel m0c + v + ((L>=16)?8:0); col = lane%16.
// Fused BN (y = x*s + t) + ReLU + bf16 pack, one b128 LDS store per lane.
__device__ __forceinline__ void store_tile(__bf16* __restrict__ dst, int stride,
                                           int nt, int m0c, int lane, v8f acc,
                                           const float* __restrict__ s,
                                           const float* __restrict__ t) {
  const int col   = nt * 16 + (lane & 15);
  const int cbase = m0c + ((lane >> 4) << 3);
  v8bf p;
#pragma unroll
  for (int v = 0; v < 8; ++v) {
    const int ch = cbase + v;
    float y = fmaxf(acc[v] * s[ch] + t[ch], 0.f);
    p[v] = (__bf16)y;
  }
  *(v8bf*)(dst + col * stride + cbase) = p;
}

// One wave computes a 16-channel strip over all 128 edges, streaming stores.
template <int KT>
__device__ __forceinline__ void gemm_stream(const __bf16* __restrict__ W, int C, int m0c,
                                            const __bf16* __restrict__ src, int sstr,
                                            __bf16* __restrict__ dst, int dstr,
                                            const float* __restrict__ s,
                                            const float* __restrict__ t, int lane) {
  v16bf a[KT];
#pragma unroll
  for (int kt = 0; kt < KT; ++kt) a[kt] = load_atile(W, C, m0c, kt, lane);
#pragma unroll
  for (int nt = 0; nt < 8; ++nt) {
    v8f acc = {};
#pragma unroll
    for (int kt = 0; kt < KT; ++kt) {
      v16bf b = load_btile(src, sstr, nt, kt * 32, lane);
      acc = __builtin_amdgcn_wmma_f32_16x16x32_bf16(false, a[kt], false, b,
                                                    (short)0, acc, false, false);
    }
    store_tile(dst, dstr, nt, m0c, lane, acc, s, t);
  }
}

// ---------------- prep: weights -> bf16, BN -> folded scale/bias ----------------
__global__ void edgeconv_prep(const float* __restrict__ w0, const float* __restrict__ w1,
                              const float* __restrict__ w2, const float* __restrict__ wa0,
                              const float* __restrict__ g0, const float* __restrict__ bb0,
                              const float* __restrict__ m0, const float* __restrict__ v0,
                              const float* __restrict__ g1, const float* __restrict__ bb1,
                              const float* __restrict__ m1, const float* __restrict__ v1,
                              const float* __restrict__ g2, const float* __restrict__ bb2,
                              const float* __restrict__ m2, const float* __restrict__ v2,
                              const float* __restrict__ ga, const float* __restrict__ bba,
                              const float* __restrict__ ma, const float* __restrict__ va,
                              const float* __restrict__ gsc, const float* __restrict__ bbsc,
                              const float* __restrict__ msc, const float* __restrict__ vsc,
                              unsigned char* __restrict__ ws) {
  const int t = blockIdx.x * blockDim.x + threadIdx.x;
  const int stride = gridDim.x * blockDim.x;
  __bf16* w0b  = (__bf16*)(ws + W0_OFF);
  __bf16* w1b  = (__bf16*)(ws + W1_OFF);
  __bf16* w2b  = (__bf16*)(ws + W2_OFF);
  __bf16* wa0b = (__bf16*)(ws + WA0_OFF);
  for (int i = t; i < 64 * 128; i += stride) w0b[i]  = (__bf16)w0[i];
  for (int i = t; i < 64 * 64;  i += stride) w1b[i]  = (__bf16)w1[i];
  for (int i = t; i < 128 * 64; i += stride) w2b[i]  = (__bf16)w2[i];
  for (int i = t; i < 64 * 128; i += stride) wa0b[i] = (__bf16)wa0[i];
  float* s0 = (float*)(ws + S0_OFF);  float* b0 = (float*)(ws + B0_OFF);
  float* s1 = (float*)(ws + S1_OFF);  float* b1 = (float*)(ws + B1_OFF);
  float* s2 = (float*)(ws + S2_OFF);  float* b2 = (float*)(ws + B2_OFF);
  float* sa = (float*)(ws + SA_OFF);  float* ba = (float*)(ws + BA_OFF);
  float* ss = (float*)(ws + SSC_OFF); float* bs = (float*)(ws + BSC_OFF);
  if (t < 64)  { float s = g0[t] * rsqrtf(v0[t] + 1e-5f); s0[t] = s; b0[t] = bb0[t] - m0[t] * s; }
  if (t < 64)  { float s = g1[t] * rsqrtf(v1[t] + 1e-5f); s1[t] = s; b1[t] = bb1[t] - m1[t] * s; }
  if (t < 128) { float s = g2[t] * rsqrtf(v2[t] + 1e-5f); s2[t] = s; b2[t] = bb2[t] - m2[t] * s; }
  if (t < 64)  { float s = ga[t] * rsqrtf(va[t] + 1e-5f); sa[t] = s; ba[t] = bba[t] - ma[t] * s; }
  if (t < 128) { float s = gsc[t] * rsqrtf(vsc[t] + 1e-5f); ss[t] = s; bs[t] = bbsc[t] - msc[t] * s; }
}

// ---------------- main fused kernel: one WG per (b, n) ----------------
__global__ __launch_bounds__(128)
void edgeconv_main(const float* __restrict__ features,
                   const float* __restrict__ wa1, const float* __restrict__ ba1,
                   const float* __restrict__ wsc,
                   const unsigned char* __restrict__ ws,
                   float* __restrict__ out) {
  __shared__ __bf16 bufA[NPTS * ASTR];   // Xcat, later Y2   (128 ch)
  __shared__ __bf16 bufB[NPTS * BSTR];   // Y0/Y1/att-hidden (64 ch)
  __shared__ float  xiF[CIN];
  __shared__ float  lg[NPTS];
  __shared__ float  att[NPTS];

  const int n    = blockIdx.x;
  const int b    = blockIdx.y;
  const int tid  = threadIdx.x;
  const int lane = tid & 31;
  const int wv   = tid >> 5;

  const __bf16* w0b  = (const __bf16*)(ws + W0_OFF);
  const __bf16* w1b  = (const __bf16*)(ws + W1_OFF);
  const __bf16* w2b  = (const __bf16*)(ws + W2_OFF);
  const __bf16* wa0b = (const __bf16*)(ws + WA0_OFF);
  const float* s0 = (const float*)(ws + S0_OFF);  const float* t0 = (const float*)(ws + B0_OFF);
  const float* s1 = (const float*)(ws + S1_OFF);  const float* t1 = (const float*)(ws + B1_OFF);
  const float* s2 = (const float*)(ws + S2_OFF);  const float* t2 = (const float*)(ws + B2_OFF);
  const float* sa = (const float*)(ws + SA_OFF);  const float* ta = (const float*)(ws + BA_OFF);
  const float* ss = (const float*)(ws + SSC_OFF); const float* ts = (const float*)(ws + BSC_OFF);

  // ---- stage 1: build Xcat = [x_i ; x_j - x_i] in LDS, bf16, [edge][ch] ----
  const float* F = features + (size_t)b * CIN * NPTS;
  {
    const int k = tid;                              // edge id (127 = zero-pad slot)
    const int j = k + (k >= n ? 1 : 0);             // gathered neighbour index
    const bool valid = (k < NPTS - 1);
    if (tid < CIN) xiF[tid] = F[tid * NPTS + n];
#pragma unroll
    for (int c0 = 0; c0 < CIN; c0 += 8) {
      v8bf xi8, xd8;
#pragma unroll
      for (int cc = 0; cc < 8; ++cc) {
        float xiv = F[(c0 + cc) * NPTS + n];
        float xjv = valid ? F[(c0 + cc) * NPTS + j] : xiv;
        xi8[cc] = (__bf16)xiv;
        xd8[cc] = (__bf16)(xjv - xiv);
      }
      *(v8bf*)(bufA + k * ASTR + c0)       = xi8;
      *(v8bf*)(bufA + k * ASTR + CIN + c0) = xd8;
    }
  }
  __syncthreads();

  // ---- GEMM0: (64x128) @ Xcat -> Y0 in bufB ----
  gemm_stream<4>(w0b, 128, wv * 16, bufA, ASTR, bufB, BSTR, s0, t0, lane);
  __syncthreads();

  // ---- GEMM1: (64x64) @ Y0 -> Y1, in-place on bufB (register-staged) ----
  {
    v16bf a0 = load_atile(w1b, 64, wv * 16, 0, lane);
    v16bf a1 = load_atile(w1b, 64, wv * 16, 1, lane);
    v8f accs[8];
#pragma unroll
    for (int nt = 0; nt < 8; ++nt) {
      v8f acc = {};
      v16bf bt = load_btile(bufB, BSTR, nt, 0, lane);
      acc = __builtin_amdgcn_wmma_f32_16x16x32_bf16(false, a0, false, bt, (short)0, acc, false, false);
      bt = load_btile(bufB, BSTR, nt, 32, lane);
      acc = __builtin_amdgcn_wmma_f32_16x16x32_bf16(false, a1, false, bt, (short)0, acc, false, false);
      accs[nt] = acc;
    }
    __syncthreads();                                // all reads of Y0 done
#pragma unroll
    for (int nt = 0; nt < 8; ++nt)
      store_tile(bufB, BSTR, nt, wv * 16, lane, accs[nt], s1, t1);
  }
  __syncthreads();

  // ---- GEMM2: (128x64) @ Y1 -> Y2 in bufA (each wave: 2 channel strips) ----
  gemm_stream<2>(w2b, 64, wv * 16,      bufB, BSTR, bufA, ASTR, s2, t2, lane);
  gemm_stream<2>(w2b, 64, 64 + wv * 16, bufB, BSTR, bufA, ASTR, s2, t2, lane);
  __syncthreads();

  // ---- attention hidden: (64x128) @ Y2 -> bufB ----
  gemm_stream<4>(wa0b, 128, wv * 16, bufA, ASTR, bufB, BSTR, sa, ta, lane);
  __syncthreads();

  // ---- logits + softmax over 127 valid edges ----
  {
    float logit = -1e30f;
    if (tid < NPTS - 1) {
      float s = 0.f;
#pragma unroll 8
      for (int c = 0; c < 64; ++c) s += wa1[c] * (float)bufB[tid * BSTR + c];
      logit = s + ba1[0];
    }
    lg[tid] = logit;
  }
  __syncthreads();
  float mx = -1e30f;
  for (int k = 0; k < NPTS - 1; ++k) mx = fmaxf(mx, lg[k]);
  att[tid] = (tid < NPTS - 1) ? __expf(lg[tid] - mx) : 0.f;
  __syncthreads();
  float sum = 0.f;
  for (int k = 0; k < NPTS - 1; ++k) sum += att[k];
  const float inv = 1.f / sum;

  // ---- attention pool (thread = output channel) + shortcut + final ReLU ----
  {
    const int o = tid;                              // 0..127
    float fts = 0.f;
    for (int k = 0; k < NPTS - 1; ++k) fts += att[k] * (float)bufA[k * ASTR + o];
    fts *= inv;
    float sc = 0.f;
#pragma unroll 8
    for (int c = 0; c < CIN; ++c) sc += wsc[o * CIN + c] * xiF[c];
    sc = sc * ss[o] + ts[o];
    out[((size_t)b * 128 + o) * NPTS + n] = fmaxf(sc + fts, 0.f);
  }
}

// ---------------- launch ----------------
extern "C" void kernel_launch(void* const* d_in, const int* in_sizes, int n_in,
                              void* d_out, int out_size, void* d_ws, size_t ws_size,
                              hipStream_t stream) {
  const float* features = (const float*)d_in[0];
  const float* w0  = (const float*)d_in[1];
  const float* g0  = (const float*)d_in[2];
  const float* bb0 = (const float*)d_in[3];
  const float* m0  = (const float*)d_in[4];
  const float* v0  = (const float*)d_in[5];
  const float* w1  = (const float*)d_in[6];
  const float* g1  = (const float*)d_in[7];
  const float* bb1 = (const float*)d_in[8];
  const float* m1  = (const float*)d_in[9];
  const float* v1  = (const float*)d_in[10];
  const float* w2  = (const float*)d_in[11];
  const float* g2  = (const float*)d_in[12];
  const float* bb2 = (const float*)d_in[13];
  const float* m2  = (const float*)d_in[14];
  const float* v2  = (const float*)d_in[15];
  const float* wa0 = (const float*)d_in[16];
  const float* ga  = (const float*)d_in[17];
  const float* ba  = (const float*)d_in[18];
  const float* ma  = (const float*)d_in[19];
  const float* va  = (const float*)d_in[20];
  const float* wa1 = (const float*)d_in[21];
  const float* ba1 = (const float*)d_in[22];
  const float* wsc = (const float*)d_in[23];
  const float* gsc = (const float*)d_in[24];
  const float* bsc = (const float*)d_in[25];
  const float* msc = (const float*)d_in[26];
  const float* vsc = (const float*)d_in[27];

  edgeconv_prep<<<64, 256, 0, stream>>>(w0, w1, w2, wa0,
                                        g0, bb0, m0, v0,
                                        g1, bb1, m1, v1,
                                        g2, bb2, m2, v2,
                                        ga, ba, ma, va,
                                        gsc, bsc, msc, vsc,
                                        (unsigned char*)d_ws);

  edgeconv_main<<<dim3(NPTS, BATCH), 128, 0, stream>>>(
      features, wa1, ba1, wsc, (const unsigned char*)d_ws, (float*)d_out);
}
